// mLSTMBlockONNXCompatible_76974403879314
// MI455X (gfx1250) — compile-verified, matrix operand order checked by reading, not run on hardware
//
#include <hip/hip_runtime.h>
#include <hip/hip_bf16.h>

// ---------------- problem constants ----------------
#define B_    16384
#define IN_   512
#define HID_  512
#define PROJ_ 1024
#define NH_   8
#define HS_   64

typedef __attribute__((ext_vector_type(16))) __bf16 v16bf;
typedef __attribute__((ext_vector_type(8)))  __bf16 v8bf;
typedef __attribute__((ext_vector_type(8)))  float  v8f;

static __device__ __forceinline__ float sigf(float x) { return 1.0f / (1.0f + expf(-x)); }

static __device__ __forceinline__ v16bf cat16(v8bf lo, v8bf hi) {
    return __builtin_shufflevector(lo, hi, 0, 1, 2, 3, 4, 5, 6, 7, 8, 9, 10, 11, 12, 13, 14, 15);
}

// ---------------- small utility kernels ----------------
__global__ void k_cvt_bf16(const float* __restrict__ s, __bf16* __restrict__ d, int n) {
    int i = blockIdx.x * blockDim.x + threadIdx.x;
    if (i < n) d[i] = (__bf16)s[i];
}

// Expand block-diagonal [NH, PROJ/NH, HS] weights into dense [PROJ, HID] bf16 (scale folds 1/sqrt(HS) for K)
__global__ void k_expand_blockdiag(const float* __restrict__ W, __bf16* __restrict__ D, float scale) {
    int i = blockIdx.x * blockDim.x + threadIdx.x;          // over PROJ_*HID_
    if (i >= PROJ_ * HID_) return;
    int d = i / HID_, e = i - d * HID_;
    int hd = d >> 7, he = e >> 6;
    float v = 0.0f;
    if (hd == he) v = W[(hd * 128 + (d & 127)) * 64 + (e & 63)] * scale;
    D[i] = (__bf16)v;
}

// LayerNorm over 512 cols, one wave per row, output bf16
__global__ __launch_bounds__(256)
void k_ln_rows(const float* __restrict__ X, const float* __restrict__ g,
               const float* __restrict__ b, __bf16* __restrict__ O) {
    int lane = threadIdx.x & 31;
    int row  = blockIdx.x * 8 + (threadIdx.x >> 5);
    const float* xr = X + (size_t)row * IN_;
    float vals[16], s = 0.f, s2 = 0.f;
#pragma unroll
    for (int j = 0; j < 16; ++j) { float v = xr[lane + 32 * j]; vals[j] = v; s += v; s2 += v * v; }
#pragma unroll
    for (int o = 16; o; o >>= 1) { s += __shfl_xor(s, o); s2 += __shfl_xor(s2, o); }
    float mu = s * (1.0f / IN_);
    float rs = rsqrtf(s2 * (1.0f / IN_) - mu * mu + 1e-5f);
#pragma unroll
    for (int j = 0; j < 16; ++j) {
        int c = lane + 32 * j;
        O[(size_t)row * IN_ + c] = (__bf16)((vals[j] - mu) * rs * g[c] + b[c]);
    }
}

// causal conv1d(K=4) + SiLU on [B, PROJ]; also emits bf16 copy of x_up_left
__global__ void k_conv_silu(const float* __restrict__ XL, const float* __restrict__ w,
                            const float* __restrict__ cb, __bf16* __restrict__ XC,
                            __bf16* __restrict__ XLb) {
    int i = blockIdx.x * blockDim.x + threadIdx.x;          // over B_*PROJ_
    if (i >= B_ * PROJ_) return;
    int c = i & (PROJ_ - 1);
    float y = cb[0];
#pragma unroll
    for (int j = 0; j < 4; ++j) {
        int cc = c + j - 3;
        if (cc >= 0) y += w[j] * XL[i + (cc - c)];
    }
    y = y * sigf(y);
    XC[i]  = (__bf16)y;
    XLb[i] = (__bf16)XL[i];
}

// ---------------- generic bf16 WMMA GEMM ----------------
// C[M,N] = act(A[M,K] @ W[K,N] + bscale*bias) (+resid); 128x128 tile, 8 waves of 32x64,
// LDS ping-pong, B stored K-transposed so all fragment reads are 2x ds_load_b128.
template<int ACT, bool BIAS, bool RESID, bool CB>
__global__ __launch_bounds__(256)
void k_gemm_bf16(const __bf16* __restrict__ A, const __bf16* __restrict__ W,
                 const float* __restrict__ bias, float bscale,
                 float* __restrict__ C, __bf16* __restrict__ Cb,
                 const float* __restrict__ resid, int N, int Kd) {
    __shared__ alignas(16) __bf16 As[2][128][40];   // [m][k], row 80B (16B aligned)
    __shared__ alignas(16) __bf16 Bt[2][128][40];   // [n][k], transposed at store time

    const int tid  = threadIdx.x;
    const int lane = tid & 31, wave = tid >> 5;
    const int wm   = wave >> 1, wn = wave & 1;
    const int hlf  = lane >> 4, ml = lane & 15;
    const int bm   = blockIdx.y * 128, bn = blockIdx.x * 128;

    v8f acc[2][4];
#pragma unroll
    for (int t0 = 0; t0 < 2; ++t0)
#pragma unroll
        for (int t1 = 0; t1 < 4; ++t1)
#pragma unroll
            for (int r = 0; r < 8; ++r) acc[t0][t1][r] = 0.0f;

    const int ar = tid >> 1, ac = (tid & 1) * 16;   // A tile: 2 threads/row, 16 elems each
    const int br = tid >> 3, bc = (tid & 7) * 16;   // B tile: 8 threads/row
    const int nIter = Kd >> 5;

    uint4 a0, a1, b0, b1;
    auto fetch = [&](int it) {
        const __bf16* sa = A + (size_t)(bm + ar) * Kd + it * 32 + ac;
        a0 = *(const uint4*)sa;
        a1 = *(const uint4*)(sa + 8);
        const __bf16* sb = W + (size_t)(it * 32 + br) * N + bn + bc;
        b0 = *(const uint4*)sb;
        b1 = *(const uint4*)(sb + 8);
    };
    auto stage = [&](int slot) {
        *(uint4*)&As[slot][ar][ac]     = a0;
        *(uint4*)&As[slot][ar][ac + 8] = a1;
        __bf16 tmp[16];
        *(uint4*)&tmp[0] = b0;
        *(uint4*)&tmp[8] = b1;
#pragma unroll
        for (int j = 0; j < 16; ++j) Bt[slot][bc + j][br] = tmp[j];   // transpose into [n][k]
    };

    fetch(0);
    stage(0);
    __syncthreads();

    for (int it = 0; it < nIter; ++it) {
        const int cur = it & 1;
        if (it + 1 < nIter) fetch(it + 1);          // global loads issue early

        v16bf af[2], bfv[4];
#pragma unroll
        for (int tm = 0; tm < 2; ++tm) {
            int m = wm * 32 + tm * 16 + ml;
            af[tm] = cat16(*(const v8bf*)&As[cur][m][8 * hlf],
                           *(const v8bf*)&As[cur][m][16 + 8 * hlf]);
        }
#pragma unroll
        for (int tn = 0; tn < 4; ++tn) {
            int n = wn * 64 + tn * 16 + ml;
            bfv[tn] = cat16(*(const v8bf*)&Bt[cur][n][8 * hlf],
                            *(const v8bf*)&Bt[cur][n][16 + 8 * hlf]);
        }
#pragma unroll
        for (int tm = 0; tm < 2; ++tm)
#pragma unroll
            for (int tn = 0; tn < 4; ++tn)
                acc[tm][tn] = __builtin_amdgcn_wmma_f32_16x16x32_bf16(
                    false, af[tm], false, bfv[tn], (short)0, acc[tm][tn], false, false);

        if (it + 1 < nIter) stage((it + 1) & 1);    // store staged regs into other buffer
        __syncthreads();
    }

#pragma unroll
    for (int tm = 0; tm < 2; ++tm)
#pragma unroll
        for (int tn = 0; tn < 4; ++tn) {
            int col = bn + wn * 64 + tn * 16 + ml;
            float bv = BIAS ? bscale * bias[col] : 0.0f;
#pragma unroll
            for (int r = 0; r < 8; ++r) {
                int row = bm + wm * 32 + tm * 16 + 8 * hlf + r;
                float v = acc[tm][tn][r] + bv;
                if (ACT == 1) v = sigf(v);
                size_t o = (size_t)row * N + col;
                if (RESID) v += resid[o];
                C[o] = v;
                if (CB) Cb[o] = (__bf16)v;
            }
        }
}

// ---------------- split-K  G_part[z] = n[zchunk]^T @ q[zchunk]  (deterministic) ----------------
// Both operands are [K=batch, 512]; both tiles stored transposed in LDS -> all frag reads b128.
__global__ __launch_bounds__(256)
void k_gemm_ntq(const __bf16* __restrict__ Nb, const __bf16* __restrict__ Qb,
                float* __restrict__ Gp, int kchunk) {
    __shared__ alignas(16) __bf16 nt[2][128][40];   // [d][k]
    __shared__ alignas(16) __bf16 qt[2][128][40];   // [e][k]
    const int tid  = threadIdx.x;
    const int lane = tid & 31, wave = tid >> 5;
    const int wm   = wave >> 1, wn = wave & 1;
    const int hlf  = lane >> 4, ml = lane & 15;
    const int bm   = blockIdx.y * 128, bn = blockIdx.x * 128;
    const int kbase = blockIdx.z * kchunk;

    v8f acc[2][4];
#pragma unroll
    for (int t0 = 0; t0 < 2; ++t0)
#pragma unroll
        for (int t1 = 0; t1 < 4; ++t1)
#pragma unroll
            for (int r = 0; r < 8; ++r) acc[t0][t1][r] = 0.0f;

    const int br = tid >> 3, bc = (tid & 7) * 16;
    const int nIter = kchunk >> 5;

    uint4 n0, n1, q0, q1;
    auto fetch = [&](int it) {
        const __bf16* sn = Nb + (size_t)(kbase + it * 32 + br) * HID_ + bm + bc;
        n0 = *(const uint4*)sn;
        n1 = *(const uint4*)(sn + 8);
        const __bf16* sq = Qb + (size_t)(kbase + it * 32 + br) * HID_ + bn + bc;
        q0 = *(const uint4*)sq;
        q1 = *(const uint4*)(sq + 8);
    };
    auto stage = [&](int slot) {
        __bf16 tn_[16], tq[16];
        *(uint4*)&tn_[0] = n0; *(uint4*)&tn_[8] = n1;
        *(uint4*)&tq[0]  = q0; *(uint4*)&tq[8]  = q1;
#pragma unroll
        for (int j = 0; j < 16; ++j) {
            nt[slot][bc + j][br] = tn_[j];
            qt[slot][bc + j][br] = tq[j];
        }
    };

    fetch(0);
    stage(0);
    __syncthreads();

    for (int it = 0; it < nIter; ++it) {
        const int cur = it & 1;
        if (it + 1 < nIter) fetch(it + 1);

        v16bf af[2], bfv[4];
#pragma unroll
        for (int tm = 0; tm < 2; ++tm) {
            int m = wm * 32 + tm * 16 + ml;
            af[tm] = cat16(*(const v8bf*)&nt[cur][m][8 * hlf],
                           *(const v8bf*)&nt[cur][m][16 + 8 * hlf]);
        }
#pragma unroll
        for (int tn = 0; tn < 4; ++tn) {
            int n = wn * 64 + tn * 16 + ml;
            bfv[tn] = cat16(*(const v8bf*)&qt[cur][n][8 * hlf],
                            *(const v8bf*)&qt[cur][n][16 + 8 * hlf]);
        }
#pragma unroll
        for (int tm = 0; tm < 2; ++tm)
#pragma unroll
            for (int tn = 0; tn < 4; ++tn)
                acc[tm][tn] = __builtin_amdgcn_wmma_f32_16x16x32_bf16(
                    false, af[tm], false, bfv[tn], (short)0, acc[tm][tn], false, false);

        if (it + 1 < nIter) stage((it + 1) & 1);
        __syncthreads();
    }

    float* G = Gp + (size_t)blockIdx.z * HID_ * HID_;
#pragma unroll
    for (int tm = 0; tm < 2; ++tm)
#pragma unroll
        for (int tn = 0; tn < 4; ++tn) {
            int col = bn + wn * 64 + tn * 16 + ml;
#pragma unroll
            for (int r = 0; r < 8; ++r) {
                int row = bm + wm * 32 + tm * 16 + 8 * hlf + r;
                G[(size_t)row * HID_ + col] = acc[tm][tn][r];
            }
        }
}

// denom[d] = max_e | sum_z Gp[z][d][e] |   (wave per row)
__global__ __launch_bounds__(256)
void k_denom_reduce(const float* __restrict__ Gp, float* __restrict__ dnm, int parts) {
    int lane = threadIdx.x & 31;
    int row  = blockIdx.x * 8 + (threadIdx.x >> 5);
    float mx = 0.0f;
#pragma unroll 4
    for (int j = 0; j < 16; ++j) {
        int e = lane + 32 * j;
        float s = 0.0f;
        for (int z = 0; z < parts; ++z) s += Gp[((size_t)z * HID_ + row) * HID_ + e];
        mx = fmaxf(mx, fabsf(s));
    }
#pragma unroll
    for (int o = 16; o; o >>= 1) mx = fmaxf(mx, __shfl_xor(mx, o));
    if (lane == 0) dnm[row] = mx;
}

// exp-stabilized gating; writes c_t/n_t/m_t into d_out and n_t as bf16 scratch
__global__ void k_gate(const float* __restrict__ I, const float* __restrict__ F,
                       const float* __restrict__ V, const float* __restrict__ Kk,
                       const float* __restrict__ cp, const float* __restrict__ np,
                       const float* __restrict__ mp,
                       float* __restrict__ Ct, float* __restrict__ Nt, float* __restrict__ Mt,
                       __bf16* __restrict__ Nb) {
    size_t i = (size_t)blockIdx.x * blockDim.x + threadIdx.x;
    if (i >= (size_t)B_ * HID_) return;
    float it = I[i], ft = F[i], m0 = mp[i];
    float mt = fmaxf(ft + m0, it);
    float ig = expf(it - mt);
    float fg = expf(ft + m0 - mt);
    float kv = Kk[i];
    float ct = fg * cp[i] + ig * (V[i] * kv);
    float nt = fg * np[i] + ig * kv;
    Ct[i] = ct; Nt[i] = nt; Mt[i] = mt;
    Nb[i] = (__bf16)nt;
}

// h_t = o*c_t*q/denom ; out = (LN(h_t)+x_skip)*silu(x_up_right); wave per row
__global__ __launch_bounds__(256)
void k_hout(const float* __restrict__ O, const float* __restrict__ CT,
            const float* __restrict__ Q, const float* __restrict__ dnm,
            const float* __restrict__ gg, const float* __restrict__ gb,
            const float* __restrict__ XS, const float* __restrict__ XUR,
            float* __restrict__ Hout, __bf16* __restrict__ OutB) {
    int lane = threadIdx.x & 31;
    int row  = blockIdx.x * 8 + (threadIdx.x >> 5);
    size_t base = (size_t)row * HID_;
    float hv[16], s = 0.f, s2 = 0.f;
#pragma unroll
    for (int j = 0; j < 16; ++j) {
        int c = lane + 32 * j; size_t i = base + c;
        float h = O[i] * CT[i] * Q[i] / dnm[c];
        hv[j] = h; Hout[i] = h; s += h; s2 += h * h;
    }
#pragma unroll
    for (int o = 16; o; o >>= 1) { s += __shfl_xor(s, o); s2 += __shfl_xor(s2, o); }
    float mu = s * (1.0f / HID_);
    float rs = rsqrtf(s2 * (1.0f / HID_) - mu * mu + 1e-5f);
#pragma unroll
    for (int j = 0; j < 16; ++j) {
        int c = lane + 32 * j; size_t i = base + c;
        float ln = (hv[j] - mu) * rs * gg[c] + gb[c];
        float xr = XUR[i];
        OutB[i] = (__bf16)((ln + XS[i]) * (xr * sigf(xr)));
    }
}

// ---------------- launch ----------------
extern "C" void kernel_launch(void* const* d_in, const int* in_sizes, int n_in,
                              void* d_out, int out_size, void* d_ws, size_t ws_size,
                              hipStream_t stream) {
    const float* x      = (const float*)d_in[0];
    const float* c_prev = (const float*)d_in[2];
    const float* n_prev = (const float*)d_in[3];
    const float* m_prev = (const float*)d_in[4];
    const float* ln_g   = (const float*)d_in[5];
    const float* ln_b   = (const float*)d_in[6];
    const float* W_upl  = (const float*)d_in[7];
    const float* b_upl  = (const float*)d_in[8];
    const float* W_upr  = (const float*)d_in[9];
    const float* b_upr  = (const float*)d_in[10];
    const float* conv_w = (const float*)d_in[11];
    const float* conv_b = (const float*)d_in[12];
    const float* W_skip = (const float*)d_in[13];
    const float* b_skip = (const float*)d_in[14];
    const float* Wq     = (const float*)d_in[15];
    const float* bq     = (const float*)d_in[16];
    const float* Wk     = (const float*)d_in[17];
    const float* bk     = (const float*)d_in[18];
    const float* Wv     = (const float*)d_in[19];
    const float* bv     = (const float*)d_in[20];
    const float* Wi     = (const float*)d_in[21];
    const float* bi     = (const float*)d_in[22];
    const float* Wf     = (const float*)d_in[23];
    const float* bfp    = (const float*)d_in[24];
    const float* Wo     = (const float*)d_in[25];
    const float* bo     = (const float*)d_in[26];
    const float* gn_g   = (const float*)d_in[27];
    const float* gn_b   = (const float*)d_in[28];
    const float* W_down = (const float*)d_in[29];
    const float* b_down = (const float*)d_in[30];

    float* out_final = (float*)d_out;
    float* out_h = out_final + (size_t)B_ * HID_;
    float* out_c = out_h + (size_t)B_ * HID_;
    float* out_n = out_c + (size_t)B_ * HID_;
    float* out_m = out_n + (size_t)B_ * HID_;

    // scratch carving (256B aligned)
    char* p = (char*)d_ws;
    auto alloc = [&](size_t bytes) -> void* {
        void* r = (void*)p;
        p += (bytes + 255) & ~(size_t)255;
        return r;
    };
    const size_t BH = (size_t)B_ * HID_, BP = (size_t)B_ * PROJ_;
    __bf16* xnorm = (__bf16*)alloc((size_t)B_ * IN_ * 2);
    float*  xulf  = (float*) alloc(BP * 4);
    __bf16* xulb  = (__bf16*)alloc(BP * 2);
    float*  xurf  = (float*) alloc(BH * 4);
    __bf16* xconv = (__bf16*)alloc(BP * 2);
    float*  xskip = (float*) alloc(BH * 4);
    float*  itld  = (float*) alloc(BH * 4);
    float*  ftld  = (float*) alloc(BH * 4);
    float*  ovals = (float*) alloc(BH * 4);
    float*  qf    = (float*) alloc(BH * 4);
    __bf16* qb    = (__bf16*)alloc(BH * 2);
    float*  kkf   = (float*) alloc(BH * 4);
    float*  vf    = (float*) alloc(BH * 4);
    __bf16* nb    = (__bf16*)alloc(BH * 2);
    __bf16* outb  = (__bf16*)alloc(BH * 2);
    const int SPLITS = 32;
    float*  Gpart = (float*) alloc((size_t)SPLITS * HID_ * HID_ * 4);
    float*  dnm   = (float*) alloc(HID_ * 4);
    __bf16* wupl  = (__bf16*)alloc((size_t)IN_ * PROJ_ * 2);
    __bf16* wupr  = (__bf16*)alloc((size_t)IN_ * HID_ * 2);
    __bf16* wskp  = (__bf16*)alloc((size_t)PROJ_ * HID_ * 2);
    __bf16* wi    = (__bf16*)alloc((size_t)PROJ_ * HID_ * 2);
    __bf16* wf    = (__bf16*)alloc((size_t)PROJ_ * HID_ * 2);
    __bf16* wo    = (__bf16*)alloc((size_t)PROJ_ * HID_ * 2);
    __bf16* wqd   = (__bf16*)alloc((size_t)PROJ_ * HID_ * 2);
    __bf16* wkd   = (__bf16*)alloc((size_t)PROJ_ * HID_ * 2);
    __bf16* wvd   = (__bf16*)alloc((size_t)PROJ_ * HID_ * 2);
    __bf16* wdn   = (__bf16*)alloc((size_t)HID_ * IN_ * 2);

    const int T = 256;
    auto cdiv = [](int a, int b) { return (a + b - 1) / b; };

    // 1) weight prep
    k_cvt_bf16<<<cdiv(IN_ * PROJ_, T), T, 0, stream>>>(W_upl, wupl, IN_ * PROJ_);
    k_cvt_bf16<<<cdiv(IN_ * HID_, T), T, 0, stream>>>(W_upr, wupr, IN_ * HID_);
    k_cvt_bf16<<<cdiv(PROJ_ * HID_, T), T, 0, stream>>>(W_skip, wskp, PROJ_ * HID_);
    k_cvt_bf16<<<cdiv(PROJ_ * HID_, T), T, 0, stream>>>(Wi, wi, PROJ_ * HID_);
    k_cvt_bf16<<<cdiv(PROJ_ * HID_, T), T, 0, stream>>>(Wf, wf, PROJ_ * HID_);
    k_cvt_bf16<<<cdiv(PROJ_ * HID_, T), T, 0, stream>>>(Wo, wo, PROJ_ * HID_);
    k_cvt_bf16<<<cdiv(HID_ * IN_, T), T, 0, stream>>>(W_down, wdn, HID_ * IN_);
    k_expand_blockdiag<<<cdiv(PROJ_ * HID_, T), T, 0, stream>>>(Wq, wqd, 1.0f);
    k_expand_blockdiag<<<cdiv(PROJ_ * HID_, T), T, 0, stream>>>(Wk, wkd, 0.125f);  // 1/sqrt(64)
    k_expand_blockdiag<<<cdiv(PROJ_ * HID_, T), T, 0, stream>>>(Wv, wvd, 1.0f);

    // 2) LN(x) -> bf16
    k_ln_rows<<<B_ / 8, T, 0, stream>>>(x, ln_g, ln_b, xnorm);

    // 3) up-projections
    k_gemm_bf16<0, true, false, false><<<dim3(PROJ_ / 128, B_ / 128), T, 0, stream>>>(
        xnorm, wupl, b_upl, 1.0f, xulf, nullptr, nullptr, PROJ_, IN_);
    k_gemm_bf16<0, true, false, false><<<dim3(HID_ / 128, B_ / 128), T, 0, stream>>>(
        xnorm, wupr, b_upr, 1.0f, xurf, nullptr, nullptr, HID_, IN_);

    // 4) conv+silu (emits x_conv bf16 and x_up_left bf16)
    k_conv_silu<<<cdiv(B_ * PROJ_, T), T, 0, stream>>>(xulf, conv_w, conv_b, xconv, xulb);

    // 5) projection GEMMs out of x_conv / x_up_left
    dim3 gP(HID_ / 128, B_ / 128);
    k_gemm_bf16<0, true, false, false><<<gP, T, 0, stream>>>(xconv, wskp, b_skip, 1.0f, xskip, nullptr, nullptr, HID_, PROJ_);
    k_gemm_bf16<0, true, false, false><<<gP, T, 0, stream>>>(xconv, wi, bi, 1.0f, itld, nullptr, nullptr, HID_, PROJ_);
    k_gemm_bf16<0, true, false, false><<<gP, T, 0, stream>>>(xconv, wf, bfp, 1.0f, ftld, nullptr, nullptr, HID_, PROJ_);
    k_gemm_bf16<1, true, false, false><<<gP, T, 0, stream>>>(xulb, wo, bo, 1.0f, ovals, nullptr, nullptr, HID_, PROJ_); // sigmoid
    k_gemm_bf16<0, true, false, true ><<<gP, T, 0, stream>>>(xconv, wqd, bq, 1.0f, qf, qb, nullptr, HID_, PROJ_);
    k_gemm_bf16<0, true, false, false><<<gP, T, 0, stream>>>(xconv, wkd, bk, 0.125f, kkf, nullptr, nullptr, HID_, PROJ_);
    k_gemm_bf16<0, true, false, false><<<gP, T, 0, stream>>>(xulb, wvd, bv, 1.0f, vf, nullptr, nullptr, HID_, PROJ_);

    // 6) gating -> c_t/n_t/m_t in d_out, n_t bf16 scratch
    k_gate<<<cdiv(B_ * HID_, T), T, 0, stream>>>(itld, ftld, vf, kkf, c_prev, n_prev, m_prev,
                                                 out_c, out_n, out_m, nb);

    // 7) denom = max|n^T q|: split-K partials then deterministic reduce
    k_gemm_ntq<<<dim3(HID_ / 128, HID_ / 128, SPLITS), T, 0, stream>>>(nb, qb, Gpart, B_ / SPLITS);
    k_denom_reduce<<<HID_ / 8, T, 0, stream>>>(Gpart, dnm, SPLITS);

    // 8) h_t + LN + skip/gate fusion -> out bf16 (and h_t into d_out)
    k_hout<<<B_ / 8, T, 0, stream>>>(ovals, out_c, qf, dnm, gn_g, gn_b, xskip, xurf, out_h, outb);

    // 9) final = out @ W_down + b_down + x
    k_gemm_bf16<0, true, true, false><<<dim3(IN_ / 128, B_ / 128), T, 0, stream>>>(
        outb, wdn, b_down, 1.0f, out_final, nullptr, x, IN_, HID_);
}
